// GRU_77025943486969
// MI455X (gfx1250) — compile-verified, matrix operand order checked by reading, not run on hardware
//
#include <hip/hip_runtime.h>
#include <hip/hip_bf16.h>

// ---------------------------------------------------------------------------
// GRU(2 layers) + FC + silu + softmax + rebalance, mixed-precision WMMA path
// for gfx1250 (MI455X).  B=256 T=128 N=H=512, 3H=1536.
// ---------------------------------------------------------------------------

#define B_  256
#define T_  128
#define N_  512
#define H_  512
#define K_  512          // contraction size for both projection GEMMs
#define G3  1536
#define LBc 0.0f
#define UBc 0.1f
#define MAX_IT 32

typedef __attribute__((ext_vector_type(16))) _Float16 v16h;
typedef __attribute__((ext_vector_type(8)))  float    v8f;

// -------------------------- f32 -> f16 conversion --------------------------
__global__ __launch_bounds__(256)
void k_cvt_f16(const float* __restrict__ in, _Float16* __restrict__ out, size_t n) {
    size_t i = (size_t)blockIdx.x * 256 + threadIdx.x;
    if (i < n) out[i] = (_Float16)in[i];
}

// ---------------------------------------------------------------------------
// GEMM: out[M][G] = A[M][512] (f16) * W[G][512]^T (f16) + bias[G], f32 accum.
// Block = 256 threads = 8 waves; block tile = 128(M) x 64(G).
// B (weights) staged cooperatively into double-buffered LDS (shared by all
// 8 waves -> 8x less global B traffic); next chunk's global load is issued
// before the current chunk's WMMAs so loads overlap compute.
// ---------------------------------------------------------------------------
__global__ __launch_bounds__(256)
void k_gemm_bias(const _Float16* __restrict__ A, const _Float16* __restrict__ W,
                 const float* __restrict__ bias, float* __restrict__ out,
                 int G)
{
    // 40-half row pitch (80 B) to spread LDS banks on the WMMA operand reads
    __shared__ __attribute__((aligned(16))) _Float16 Bs[2][64][40];

    const int tid  = threadIdx.x;
    const int lane = tid & 31;
    const int wave = tid >> 5;
    const int r    = lane & 15;
    const int kh   = lane >> 4;                 // K-half selector (0/1)
    const int m0   = blockIdx.y * 128 + wave * 16;
    const int g0   = blockIdx.x * 64;

    // cooperative B staging: each thread moves 16 B (8 halves) per chunk
    const int brow = tid >> 2;                  // 0..63
    const int bcol = (tid & 3) * 8;             // 0,8,16,24 halves
    const _Float16* Wg   = W + (size_t)(g0 + brow) * K_ + bcol;
    const _Float16* Arow = A + (size_t)(m0 + r) * K_ + kh * 16;

    uint4 stg = *(const uint4*)(Wg);            // chunk 0 of B
    v16h  a   = *(const v16h*)(Arow);           // chunk 0 of A
    *(uint4*)&Bs[0][brow][bcol] = stg;
    __syncthreads();

    v8f acc[4] = {};
    const int nk = K_ / 32;                     // 16 chunks
    for (int ks = 0; ks < nk; ++ks) {
        const int buf = ks & 1;
        uint4 stg_n = {};
        v16h  a_n   = {};
        if (ks + 1 < nk) {                      // issue next chunk's loads now
            __builtin_prefetch(Wg + (ks + 2) * 32, 0, 1);   // global_prefetch_b8
            stg_n = *(const uint4*)(Wg + (size_t)(ks + 1) * 32);
            a_n   = *(const v16h*)(Arow + (size_t)(ks + 1) * 32);
        }
#pragma unroll
        for (int j = 0; j < 4; ++j) {
            v16h b = *(const v16h*)&Bs[buf][j * 16 + r][kh * 16];
            acc[j] = __builtin_amdgcn_wmma_f32_16x16x32_f16(
                         false, a, false, b, (short)0, acc[j], false, false);
        }
        if (ks + 1 < nk) {
            *(uint4*)&Bs[buf ^ 1][brow][bcol] = stg_n;   // other buffer: no race
            a = a_n;
            __syncthreads();                              // make buf^1 visible
        }
    }

#pragma unroll
    for (int j = 0; j < 4; ++j) {
        const int g  = g0 + j * 16 + r;
        const float bv = bias[g];
#pragma unroll
        for (int rr = 0; rr < 8; ++rr)
            out[(size_t)(m0 + rr + 8 * kh) * G + g] = acc[j][rr] + bv;
    }
}

// ---------------------------------------------------------------------------
// Persistent GRU scan.  Each block owns 16 batch rows for the whole time loop;
// h kept in LDS (f32 master + f16 WMMA operand copy).  512 threads = 16 waves;
// per step each wave computes 6 of the 96 gate tiles of gh = h * W_hh^T.
// Weight tiles are register double-buffered (bcur/bnext) so the 6 WMMAs of
// step k run against registers while step k+1's global loads are in flight.
// ---------------------------------------------------------------------------
__global__ __launch_bounds__(512)
void k_gru_scan(const float* __restrict__ XI,        // [B][T][3H] (b_ih folded)
                const _Float16* __restrict__ Whh,    // [3H][H] f16
                const float* __restrict__ bhh,       // [3H]
                _Float16* __restrict__ hseq16,       // [B][T][H] f16 or null
                float* __restrict__ hT,              // [B][H] or null
                int T)
{
    __shared__ float                                  gh[16 * G3];   // 96 KB
    __shared__ __attribute__((aligned(32))) _Float16  hf[16 * H_];   // 16 KB
    __shared__ float                                  hv[16 * H_];   // 32 KB

    const int tid  = threadIdx.x;
    const int lane = tid & 31;
    const int wave = tid >> 5;          // 0..15
    const int r    = lane & 15;
    const int kh   = lane >> 4;
    const int b0   = blockIdx.x * 16;   // batch rows owned by this block
    const int grow = tid >> 5;          // gate-phase row (0..15)
    const int gl   = tid & 31;          // gate-phase lane (coalesced cols)

    // per-wave weight tile base pointers (row = gate column, contiguous K)
    const _Float16* Wt = Whh + (size_t)(wave * 6 * 16 + r) * H_ + kh * 16;

    for (int i = tid; i < 16 * H_; i += 512) { hv[i] = 0.f; hf[i] = (_Float16)0.f; }
    __syncthreads();

    for (int t = 0; t < T; ++t) {
        // ---- gh = h * Whh^T via WMMA, software-pipelined weight stream ----
        v8f acc[6] = {};
        const _Float16* hrow = hf + (size_t)r * H_ + kh * 16;
        v16h bcur[6];
#pragma unroll
        for (int j = 0; j < 6; ++j)
            bcur[j] = *(const v16h*)(Wt + (size_t)j * 16 * H_);
        for (int k = 0; k < H_; k += 32) {
            v16h a = *(const v16h*)(hrow + k);
            v16h bnext[6];
            if (k + 32 < H_) {
#pragma unroll
                for (int j = 0; j < 6; ++j)
                    bnext[j] = *(const v16h*)(Wt + (size_t)j * 16 * H_ + k + 32);
            }
#pragma unroll
            for (int j = 0; j < 6; ++j)
                acc[j] = __builtin_amdgcn_wmma_f32_16x16x32_f16(
                             false, a, false, bcur[j], (short)0, acc[j], false, false);
            if (k + 32 < H_) {
#pragma unroll
                for (int j = 0; j < 6; ++j) bcur[j] = bnext[j];
            }
        }
#pragma unroll
        for (int j = 0; j < 6; ++j) {
            const int g = (wave * 6 + j) * 16 + r;
#pragma unroll
            for (int rr = 0; rr < 8; ++rr)
                gh[(rr + 8 * kh) * G3 + g] = acc[j][rr];
        }
        __syncthreads();

        // ---- gates (coalesced: lane = contiguous columns) ----
        const float* xi = XI + ((size_t)(b0 + grow) * T + t) * G3;
#pragma unroll 4
        for (int i = 0; i < 16; ++i) {
            const int n = gl + 32 * i;
            const float hr = gh[grow * G3 + n]          + bhh[n];
            const float hz = gh[grow * G3 + H_  + n]    + bhh[H_ + n];
            const float hn = gh[grow * G3 + 2*H_ + n]   + bhh[2*H_ + n];
            const float xr = xi[n], xz = xi[H_ + n], xn = xi[2*H_ + n];
            const float rg = 1.f / (1.f + __expf(-(xr + hr)));
            const float zg = 1.f / (1.f + __expf(-(xz + hz)));
            const float e2 = __expf(2.f * (xn + rg * hn));
            const float ng = (e2 - 1.f) / (e2 + 1.f);
            const float h2 = (1.f - zg) * ng + zg * hv[grow * H_ + n];
            hv[grow * H_ + n] = h2;
            hf[grow * H_ + n] = (_Float16)h2;
            if (hseq16) hseq16[((size_t)(b0 + grow) * T + t) * H_ + n] = (_Float16)h2;
            if (hT && t == T - 1) hT[(size_t)(b0 + grow) * H_ + n] = h2;
        }
        __syncthreads();
    }
}

// ---------------------------------------------------------------------------
// Head: logits = h_T * fc_w^T + fc_b; silu; softmax; iterative rebalance.
// One block per batch row; 256 threads * 2 elements = 512 columns.
// ---------------------------------------------------------------------------
__device__ __forceinline__ float blk_sum(float v, float* red, int tid) {
    red[tid] = v; __syncthreads();
    for (int s = 128; s > 0; s >>= 1) { if (tid < s) red[tid] += red[tid + s]; __syncthreads(); }
    float x = red[0]; __syncthreads(); return x;
}
__device__ __forceinline__ float blk_max(float v, float* red, int tid) {
    red[tid] = v; __syncthreads();
    for (int s = 128; s > 0; s >>= 1) { if (tid < s) red[tid] = fmaxf(red[tid], red[tid + s]); __syncthreads(); }
    float x = red[0]; __syncthreads(); return x;
}

__global__ __launch_bounds__(256)
void k_head(const float* __restrict__ hT, const float* __restrict__ fcw,
            const float* __restrict__ fcb, float* __restrict__ out)
{
    __shared__ float red[256];
    __shared__ float hs[H_];
    const int b = blockIdx.x, tid = threadIdx.x;

    for (int i = tid; i < H_; i += 256) hs[i] = hT[(size_t)b * H_ + i];
    __syncthreads();

    float lg[2];
#pragma unroll
    for (int j = 0; j < 2; ++j) {
        const int n = 2 * tid + j;
        const float* w = fcw + (size_t)n * H_;
        float s = fcb[n];
        for (int k = 0; k < H_; ++k) s += hs[k] * w[k];
        lg[j] = s / (1.f + __expf(-s));                       // silu
    }
    // softmax
    const float mx = blk_max(fmaxf(lg[0], lg[1]), red, tid);
    float e0 = __expf(lg[0] - mx), e1 = __expf(lg[1] - mx);
    const float inv = 1.f / blk_sum(e0 + e1, red, tid);
    const float w0 = e0 * inv, w1 = e1 * inv;

    // rebalance (fixed 32 iterations, matches jax.lax.scan semantics)
    float old0 = w0, old1 = w1;
    float wc0 = fminf(fmaxf(w0, LBc), UBc), wc1 = fminf(fmaxf(w1, LBc), UBc);
    bool cont = true;
    for (int it = 0; it < MAX_IT; ++it) {
        const float leftover = blk_sum((old0 - wc0) + (old1 - wc1), red, tid);
        const float n0 = (wc0 != UBc) ? wc0 : 0.f;
        const float n1 = (wc1 != UBc) ? wc1 : 0.f;
        const float nomsum = blk_sum(n0 + n1, red, tid);
        const float g = leftover / nomsum;
        const float a0 = wc0 + g * n0, a1 = wc1 + g * n1;
        const bool again = blk_max(fmaxf(a0, a1), red, tid) > UBc;
        const float b0v = again ? fminf(fmaxf(a0, LBc), UBc) : a0;
        const float b1v = again ? fminf(fmaxf(a1, LBc), UBc) : a1;
        if (cont) { old0 = a0; old1 = a1; wc0 = b0v; wc1 = b1v; }
        cont = cont && again;
    }
    out[(size_t)b * N_ + 2 * tid + 0] = wc0;
    out[(size_t)b * N_ + 2 * tid + 1] = wc1;
}

// ---------------------------------------------------------------------------
extern "C" void kernel_launch(void* const* d_in, const int* in_sizes, int n_in,
                              void* d_out, int out_size, void* d_ws, size_t ws_size,
                              hipStream_t stream)
{
    const float* x     = (const float*)d_in[0];
    const float* Wih0  = (const float*)d_in[1];
    const float* Whh0  = (const float*)d_in[2];
    const float* bih0  = (const float*)d_in[3];
    const float* bhh0  = (const float*)d_in[4];
    const float* Wih1  = (const float*)d_in[5];
    const float* Whh1  = (const float*)d_in[6];
    const float* bih1  = (const float*)d_in[7];
    const float* bhh1  = (const float*)d_in[8];
    const float* fcw   = (const float*)d_in[9];
    const float* fcb   = (const float*)d_in[10];

    char* ws = (char*)d_ws;
    size_t off = 0;
    auto carve = [&](size_t bytes) { char* p = ws + off; off += (bytes + 255) & ~(size_t)255; return p; };

    _Float16* Xf16   = (_Float16*)carve((size_t)B_ * T_ * N_ * 2);   // 32 MB
    _Float16* Wih0h  = (_Float16*)carve((size_t)G3 * N_ * 2);
    _Float16* Whh0h  = (_Float16*)carve((size_t)G3 * H_ * 2);
    _Float16* Wih1h  = (_Float16*)carve((size_t)G3 * H_ * 2);
    _Float16* Whh1h  = (_Float16*)carve((size_t)G3 * H_ * 2);
    float*    XI     = (float*)   carve((size_t)B_ * T_ * G3 * 4);   // 192 MB (reused both layers)
    _Float16* Hseq16 = (_Float16*)carve((size_t)B_ * T_ * H_ * 2);   // 32 MB
    float*    hTf    = (float*)   carve((size_t)B_ * H_ * 4);

    auto cvt = [&](const float* src, _Float16* dst, size_t n) {
        k_cvt_f16<<<(unsigned)((n + 255) / 256), 256, 0, stream>>>(src, dst, n);
    };
    cvt(x,    Xf16,  (size_t)B_ * T_ * N_);
    cvt(Wih0, Wih0h, (size_t)G3 * N_);
    cvt(Whh0, Whh0h, (size_t)G3 * H_);
    cvt(Wih1, Wih1h, (size_t)G3 * H_);
    cvt(Whh1, Whh1h, (size_t)G3 * H_);

    const int M = B_ * T_;                      // 32768
    dim3 ggemm(G3 / 64, M / 128);               // (24, 256)

    // Layer 0: input projection + recurrent scan (emits f16 sequence)
    k_gemm_bias<<<ggemm, 256, 0, stream>>>(Xf16, Wih0h, bih0, XI, G3);
    k_gru_scan <<<B_ / 16, 512, 0, stream>>>(XI, Whh0h, bhh0, Hseq16, nullptr, T_);

    // Layer 1: input projection from layer-0 sequence + scan (emits h_T only)
    k_gemm_bias<<<ggemm, 256, 0, stream>>>(Hseq16, Wih1h, bih1, XI, G3);
    k_gru_scan <<<B_ / 16, 512, 0, stream>>>(XI, Whh1h, bhh1, nullptr, hTf, T_);

    // FC + silu + softmax + rebalance
    k_head<<<B_, 256, 0, stream>>>(hTf, fcw, fcb, (float*)d_out);
}
